// TppAngleNet_29506425323833
// MI455X (gfx1250) — compile-verified, matrix operand order checked by reading, not run on hardware
//
#include <hip/hip_runtime.h>
#include <hip/hip_bf16.h>
#include <math.h>

// ---------------------------------------------------------------------------
// WMMA types / helpers (CDNA5 wave32, V_WMMA_F32_16X16X32_F16)
// ---------------------------------------------------------------------------
typedef __attribute__((ext_vector_type(16))) _Float16 v16h;
typedef __attribute__((ext_vector_type(8)))  float    v8f;

#define WMMA_F16(a, b, c) \
  __builtin_amdgcn_wmma_f32_16x16x32_f16(false, (a), false, (b), (short)0, (c), false, false)

// Swizzled fragment storage. For a [Kp x Ncols] matrix consumed as WMMA
// operand tiles (16 rows-of-the-striped-dim x 32 K), we store each lane's 16
// f16 elements contiguously:
//   flat = ((tile16 * (Kp/32) + kc) * 32 + lane) * 16 + e
// where lane = (row%16) + 16*half and element e covers K per ISA 7.12.2:
//   kl = K%32 : half = (kl>>3)&1,  e = (kl&7) + ((kl>>4)<<3)
// A-fragments (row = M) and B-fragments (row = N) use the same striping, so
// one 256-bit load per lane yields a complete fragment.
static __device__ __forceinline__ int sw_idx(int m16, int k) {
  int kc = k >> 5, kl = k & 31;
  int half = (kl >> 3) & 1;
  int e = (kl & 7) + ((kl >> 4) << 3);
  return ((kc << 5) + (half << 4) + m16) * 16 + e;   // within one 16-wide tile
}

// A fragment from a swizzled LDS tile (single 16-row tile, KC = Kp/32 chunks)
static __device__ __forceinline__ v16h frag_a_sw(const _Float16* t, int kc) {
  int lane = threadIdx.x & 31;
  return ((const v16h*)t)[kc * 32 + lane];
}

// B fragment from swizzled global weights: tile nt (16 cols), chunk kc.
static __device__ __forceinline__ v16h frag_b_sw(const _Float16* __restrict__ W,
                                                 int KC, int kc, int nt) {
  int lane = threadIdx.x & 31;
  return ((const v16h*)W)[((size_t)nt * KC + kc) * 32 + lane];
}

// ---------------------------------------------------------------------------
// Weight convert+swizzle: f32 [rows x cols] -> swizzled f16 fragments for a
// [rowsP x cols] matrix (rowsP mult of 32, cols mult of 16), zero-padded rows.
// ---------------------------------------------------------------------------
__global__ void cvt_swizzle_f16(const float* __restrict__ src, int rows, int cols,
                                _Float16* __restrict__ dst, int rowsP) {
  int KC = rowsP >> 5;
  size_t total = (size_t)rowsP * cols;
  for (size_t o = (size_t)blockIdx.x * blockDim.x + threadIdx.x; o < total;
       o += (size_t)gridDim.x * blockDim.x) {
    int e     = (int)(o & 15);
    int lane  = (int)((o >> 4) & 31);
    int chunk = (int)(o >> 9);
    int kc = chunk % KC, nt = chunk / KC;
    int n = nt * 16 + (lane & 15);
    int kk = kc * 32 + ((e < 8) ? e : (e + 8)) + 8 * (lane >> 4);
    dst[o] = (kk < rows && n < cols) ? (_Float16)src[(size_t)kk * cols + n]
                                     : (_Float16)0.f;
  }
}

// ---------------------------------------------------------------------------
// kNN top-8 (includes self at d=0).  One wave per (i, b); grid = (N, B).
// smem: xi[C] | candD[256] | candJ[256]
// ---------------------------------------------------------------------------
__global__ void knn_topk(const float* __restrict__ X, int sx, int C, int N,
                         int* __restrict__ out_idx) {
  extern __shared__ char smem[];
  float* xi_s = (float*)smem;
  float* cd = xi_s + C;
  int* cj = (int*)(cd + 256);

  int i = blockIdx.x, b = blockIdx.y;
  int lane = threadIdx.x;

  const float* xi = X + ((size_t)b * N + i) * sx;
  for (int c = lane; c < C; c += 32) xi_s[c] = xi[c];
  __syncthreads();

  float d8[8];
  int   j8[8];
#pragma unroll
  for (int s = 0; s < 8; ++s) { d8[s] = 3.4e38f; j8[s] = 0; }

  for (int j = lane; j < N; j += 32) {
    const float* xj = X + ((size_t)b * N + j) * sx;
    float d2 = 0.f;
    for (int c = 0; c < C; ++c) { float t = xi_s[c] - xj[c]; d2 += t * t; }
    if (d2 < d8[7]) {
      d8[7] = d2; j8[7] = j;
#pragma unroll
      for (int s = 7; s > 0; --s)
        if (d8[s] < d8[s - 1]) {
          float td = d8[s]; d8[s] = d8[s - 1]; d8[s - 1] = td;
          int   tj = j8[s]; j8[s] = j8[s - 1]; j8[s - 1] = tj;
        }
    }
  }
#pragma unroll
  for (int s = 0; s < 8; ++s) { cd[lane * 8 + s] = d8[s]; cj[lane * 8 + s] = j8[s]; }
  __syncthreads();
  if (lane == 0) {
    for (int s = 0; s < 8; ++s) {
      float best = cd[0]; int bm = 0;
      for (int t = 1; t < 256; ++t)
        if (cd[t] < best) { best = cd[t]; bm = t; }
      out_idx[((size_t)b * N + i) * 8 + s] = cj[bm];
      cd[bm] = 3.4e38f;
    }
  }
}

// ---------------------------------------------------------------------------
// EdgeConv: one wave handles 2 points (16 edges); grid = (N/2, B).
// E=[xi, xj-xi] built swizzled in LDS, hidden=relu(E@W1+b1) (swizzled LDS),
// out=hidden@W2+b2, max over the 8 neighbor rows via the C/D lane-half layout.
// smem: E f16 [16 x KP] | Hid f16 [16 x Hp]   (both in fragment order)
// ---------------------------------------------------------------------------
__global__ void edgeconv_wmma(const float* __restrict__ X, int sx, int C,
                              const int* __restrict__ idx,
                              const _Float16* __restrict__ W1,
                              const float* __restrict__ B1, int H, int Hp,
                              const _Float16* __restrict__ W2,
                              const float* __restrict__ B2, int Cout,
                              float* __restrict__ Y, int sy, int N) {
  extern __shared__ char smem[];
  int KP = (((2 * C) + 31) >> 5) << 5;
  _Float16* E   = (_Float16*)smem;   // 16*KP halves, swizzled
  _Float16* Hid = E + 16 * KP;       // 16*Hp halves, swizzled

  int p0 = blockIdx.x * 2;
  int b  = blockIdx.y;
  int lane = threadIdx.x;

  __builtin_prefetch(W1, 0, 1);      // global_prefetch_b8 on the weight stream
  __builtin_prefetch(W2, 0, 1);

  for (int m = 0; m < 16; ++m) {
    int p = p0 + (m >> 3);
    int j = idx[((size_t)b * N + p) * 8 + (m & 7)];
    const float* xi = X + ((size_t)b * N + p) * sx;
    const float* xj = X + ((size_t)b * N + j) * sx;
    for (int c = lane; c < C; c += 32) {
      float vi = xi[c];
      E[sw_idx(m, c)]     = (_Float16)vi;
      E[sw_idx(m, C + c)] = (_Float16)(xj[c] - vi);
    }
    for (int c = 2 * C + lane; c < KP; c += 32) E[sw_idx(m, c)] = (_Float16)0.f;
  }
  if (Hp > H)
    for (int m = 0; m < 16; ++m)
      for (int c = H + lane; c < Hp; c += 32) Hid[sw_idx(m, c)] = (_Float16)0.f;
  __syncthreads();

  int nl = lane & 15, half = lane >> 4;
  int KC1 = KP >> 5, KC2 = Hp >> 5;

  for (int h0 = 0; h0 < H; h0 += 16) {
    v8f acc = {};
    for (int kc = 0; kc < KC1; ++kc)
      acc = WMMA_F16(frag_a_sw(E, kc), frag_b_sw(W1, KC1, kc, h0 >> 4), acc);
    float bias = B1[h0 + nl];
#pragma unroll
    for (int r = 0; r < 8; ++r) {
      float v = acc[r] + bias;
      Hid[sw_idx(r + 8 * half, h0 + nl)] = (_Float16)(v > 0.f ? v : 0.f);
    }
  }
  __syncthreads();

  for (int o0 = 0; o0 < Cout; o0 += 16) {
    v8f acc = {};
    for (int kc = 0; kc < KC2; ++kc)
      acc = WMMA_F16(frag_a_sw(Hid, kc), frag_b_sw(W2, KC2, kc, o0 >> 4), acc);
    float bias = B2[o0 + nl];
    float mx = acc[0] + bias;
#pragma unroll
    for (int r = 1; r < 8; ++r) { float v = acc[r] + bias; mx = v > mx ? v : mx; }
    // lane half=0 -> rows 0..7 (point p0), half=1 -> rows 8..15 (point p0+1)
    Y[((size_t)b * N + p0 + half) * sy + o0 + nl] = mx;
  }
}

// ---------------------------------------------------------------------------
// Two-layer point MLP (relu between, none after): 16 rows per wave.
// K must be a multiple of 32.  smem: E f16 [16xK] | Hid f16 [16xHp], swizzled.
// ---------------------------------------------------------------------------
__global__ void mlp2_wmma(const float* __restrict__ X, int sx, int K,
                          const _Float16* __restrict__ W1,
                          const float* __restrict__ B1, int H, int Hp,
                          const _Float16* __restrict__ W2,
                          const float* __restrict__ B2, int Cout,
                          float* __restrict__ Y, int sy) {
  extern __shared__ char smem[];
  _Float16* E   = (_Float16*)smem;
  _Float16* Hid = E + 16 * K;
  size_t row0 = (size_t)blockIdx.x * 16;
  int lane = threadIdx.x;

  __builtin_prefetch(W1, 0, 1);
  __builtin_prefetch(W2, 0, 1);

  for (int m = 0; m < 16; ++m) {
    const float* xr = X + (row0 + m) * sx;
    for (int c = lane; c < K; c += 32) E[sw_idx(m, c)] = (_Float16)xr[c];
  }
  if (Hp > H)
    for (int m = 0; m < 16; ++m)
      for (int c = H + lane; c < Hp; c += 32) Hid[sw_idx(m, c)] = (_Float16)0.f;
  __syncthreads();

  int nl = lane & 15, half = lane >> 4;
  int KC1 = K >> 5, KC2 = Hp >> 5;

  for (int h0 = 0; h0 < H; h0 += 16) {
    v8f acc = {};
    for (int kc = 0; kc < KC1; ++kc)
      acc = WMMA_F16(frag_a_sw(E, kc), frag_b_sw(W1, KC1, kc, h0 >> 4), acc);
    float bias = B1[h0 + nl];
#pragma unroll
    for (int r = 0; r < 8; ++r) {
      float v = acc[r] + bias;
      Hid[sw_idx(r + 8 * half, h0 + nl)] = (_Float16)(v > 0.f ? v : 0.f);
    }
  }
  __syncthreads();
  for (int o0 = 0; o0 < Cout; o0 += 16) {
    v8f acc = {};
    for (int kc = 0; kc < KC2; ++kc)
      acc = WMMA_F16(frag_a_sw(Hid, kc), frag_b_sw(W2, KC2, kc, o0 >> 4), acc);
    float bias = B2[o0 + nl];
#pragma unroll
    for (int r = 0; r < 8; ++r)
      Y[(row0 + r + 8 * half) * sy + o0 + nl] = acc[r] + bias;
  }
}

// ---------------------------------------------------------------------------
// Plain GEMM (no bias, no act): Y[MxNn] = X[MxK] @ W[KxNn].  grid (M/16, Nn/16).
// ---------------------------------------------------------------------------
__global__ void gemm_wmma(const float* __restrict__ X, int sx, int K,
                          const _Float16* __restrict__ W, int Nn,
                          float* __restrict__ Y, int sy) {
  extern __shared__ char smem[];
  _Float16* E = (_Float16*)smem;     // 16*K halves, swizzled
  size_t row0 = (size_t)blockIdx.x * 16;
  int nt = blockIdx.y;
  int lane = threadIdx.x;

  __builtin_prefetch(W, 0, 1);

  for (int m = 0; m < 16; ++m) {
    const float* xr = X + (row0 + m) * sx;
    for (int c = lane; c < K; c += 32) E[sw_idx(m, c)] = (_Float16)xr[c];
  }
  __syncthreads();
  int KC = K >> 5;
  v8f acc = {};
  for (int kc = 0; kc < KC; ++kc)
    acc = WMMA_F16(frag_a_sw(E, kc), frag_b_sw(W, KC, kc, nt), acc);
  int nl = lane & 15, half = lane >> 4;
#pragma unroll
  for (int r = 0; r < 8; ++r)
    Y[(row0 + r + 8 * half) * sy + nt * 16 + nl] = acc[r];
}

// ---------------------------------------------------------------------------
// Global max-pool over N: g[b][c] = max_p sf[b][p][c].  grid = B, block = 128.
// ---------------------------------------------------------------------------
__global__ void maxpool_kernel(const float* __restrict__ sf, float* __restrict__ g, int N) {
  int b = blockIdx.x, c = threadIdx.x;
  float mx = -3.4e38f;
  for (int p = 0; p < N; ++p) {
    float v = sf[((size_t)b * N + p) * 128 + c];
    mx = v > mx ? v : mx;
  }
  g[b * 128 + c] = mx;
}

// hg[b][c] = ec_b1[c] + sum_k g[b][k] * ec_w1[256+k][c].  1 block x 256.
__global__ void hg_kernel(const float* __restrict__ g, const float* __restrict__ ecw1,
                          const float* __restrict__ ecb1, float* __restrict__ hg) {
  int t = threadIdx.x;
  int b = t >> 7, c = t & 127;
  float acc = ecb1[c];
  for (int k = 0; k < 128; ++k)
    acc += g[b * 128 + k] * ecw1[(size_t)(256 + k) * 128 + c];
  hg[b * 128 + c] = acc;
}

// ---------------------------------------------------------------------------
// Pair classifier, fused (never materializes [B,P,128]).
// pr(i,j) = i*(2N-i-1)/2 + (j-i-1)  matches np.triu_indices(N,1) order.
// out[0 .. 2P) = sigmoid(logit) (pair_prob, [B,P]); out[2P .. 4P) = logit.
// ---------------------------------------------------------------------------
__global__ void pair_kernel(const float* __restrict__ hi, const float* __restrict__ hj,
                            const float* __restrict__ hg, const float* __restrict__ ecw2,
                            const float* __restrict__ ecb2, float* __restrict__ out, int N) {
  __shared__ float hi_s[2][128], hg_s[2][128], w2_s[128];
  int i = blockIdx.x, tid = threadIdx.x;
  if (tid < 128) w2_s[tid] = ecw2[tid];
  { int b = tid >> 7, c = tid & 127;
    hi_s[b][c] = hi[((size_t)b * N + i) * 128 + c];
    hg_s[b][c] = hg[b * 128 + c]; }
  __syncthreads();

  long P = (long)N * (N - 1) / 2;
  long base = (long)i * (2L * N - i - 1) / 2;
  float b2 = ecb2[0];
  for (int j = i + 1 + tid; j < N; j += 256) {
    long pr = base + (j - i - 1);
    for (int b = 0; b < 2; ++b) {
      const float* hjr = hj + ((size_t)b * N + j) * 128;
      float acc = b2;
      for (int c = 0; c < 128; ++c) {
        float h = hi_s[b][c] + hjr[c] + hg_s[b][c];
        acc += (h > 0.f ? h : 0.f) * w2_s[c];
      }
      out[(size_t)b * P + pr] = 1.f / (1.f + __expf(-acc));
      out[(size_t)2 * P + (size_t)b * P + pr] = acc;
    }
  }
}

// ---------------------------------------------------------------------------
// Host orchestration
// ---------------------------------------------------------------------------
extern "C" void kernel_launch(void* const* d_in, const int* in_sizes, int n_in,
                              void* d_out, int out_size, void* d_ws, size_t ws_size,
                              hipStream_t stream) {
  (void)in_sizes; (void)n_in; (void)out_size; (void)ws_size;
  const int B = 2, N = 1000;

  const float* pos   = (const float*)d_in[0];
  const float* c1w1  = (const float*)d_in[1];  const float* c1b1 = (const float*)d_in[2];
  const float* c1w2  = (const float*)d_in[3];  const float* c1b2 = (const float*)d_in[4];
  const float* c2w1  = (const float*)d_in[5];  const float* c2b1 = (const float*)d_in[6];
  const float* c2w2  = (const float*)d_in[7];  const float* c2b2 = (const float*)d_in[8];
  const float* c3w1  = (const float*)d_in[9];  const float* c3b1 = (const float*)d_in[10];
  const float* c3w2  = (const float*)d_in[11]; const float* c3b2 = (const float*)d_in[12];
  const float* smw1  = (const float*)d_in[13]; const float* smb1 = (const float*)d_in[14];
  const float* smw2  = (const float*)d_in[15]; const float* smb2 = (const float*)d_in[16];
  const float* ecw1  = (const float*)d_in[17]; const float* ecb1 = (const float*)d_in[18];
  const float* ecw2  = (const float*)d_in[19]; const float* ecb2 = (const float*)d_in[20];
  float* out = (float*)d_out;

  // bump allocator in workspace (256B aligned)
  size_t off = 0;
  auto alloc = [&](size_t bytes) -> char* {
    off = (off + 255) & ~(size_t)255;
    char* p = (char*)d_ws + off;
    off += bytes;
    return p;
  };
  int*   idx1 = (int*)  alloc((size_t)B * N * 8 * 4);
  int*   idx2 = (int*)  alloc((size_t)B * N * 8 * 4);
  int*   idx3 = (int*)  alloc((size_t)B * N * 8 * 4);
  float* xcat = (float*)alloc((size_t)B * N * 672 * 4);  // [x1 | x2 | x3]
  float* sf   = (float*)alloc((size_t)B * N * 128 * 4);
  float* g    = (float*)alloc(256 * 4);
  float* hi   = (float*)alloc((size_t)B * N * 128 * 4);
  float* hj   = (float*)alloc((size_t)B * N * 128 * 4);
  float* hg   = (float*)alloc(256 * 4);
  _Float16* c1w1p = (_Float16*)alloc((size_t)32 * 16 * 2);
  _Float16* c1w2p = (_Float16*)alloc((size_t)32 * 32 * 2);
  _Float16* c2w1p = (_Float16*)alloc((size_t)64 * 64 * 2);
  _Float16* c2w2p = (_Float16*)alloc((size_t)64 * 128 * 2);
  _Float16* c3w1p = (_Float16*)alloc((size_t)256 * 256 * 2);
  _Float16* c3w2p = (_Float16*)alloc((size_t)256 * 512 * 2);
  _Float16* smw1p = (_Float16*)alloc((size_t)672 * 256 * 2);
  _Float16* smw2p = (_Float16*)alloc((size_t)256 * 128 * 2);
  _Float16* ehiw  = (_Float16*)alloc((size_t)128 * 128 * 2);
  _Float16* ehjw  = (_Float16*)alloc((size_t)128 * 128 * 2);

  auto cvt = [&](const float* s, int r, int c, _Float16* d, int rp) {
    int total = rp * c;
    cvt_swizzle_f16<<<(total + 255) / 256, 256, 0, stream>>>(s, r, c, d, rp);
  };
  cvt(c1w1, 6, 16, c1w1p, 32);       cvt(c1w2, 16, 32, c1w2p, 32);
  cvt(c2w1, 64, 64, c2w1p, 64);      cvt(c2w2, 64, 128, c2w2p, 64);
  cvt(c3w1, 256, 256, c3w1p, 256);   cvt(c3w2, 256, 512, c3w2p, 256);
  cvt(smw1, 672, 256, smw1p, 672);   cvt(smw2, 256, 128, smw2p, 256);
  cvt(ecw1, 128, 128, ehiw, 128);    cvt(ecw1 + 128 * 128, 128, 128, ehjw, 128);

  auto knn_smem = [](int C) { return (size_t)(C + 512) * 4; };

  // --- EdgeConv 1: pos(C=3) -> xcat[:, 0:32]
  knn_topk<<<dim3(N, B), 32, knn_smem(3), stream>>>(pos, 3, 3, N, idx1);
  edgeconv_wmma<<<dim3(N / 2, B), 32, (size_t)(16 * 32 + 16 * 32) * 2, stream>>>(
      pos, 3, 3, idx1, c1w1p, c1b1, 16, 32, c1w2p, c1b2, 32, xcat + 0, 672, N);

  // --- EdgeConv 2: x1(C=32) -> xcat[:, 32:160]
  knn_topk<<<dim3(N, B), 32, knn_smem(32), stream>>>(xcat, 672, 32, N, idx2);
  edgeconv_wmma<<<dim3(N / 2, B), 32, (size_t)(16 * 64 + 16 * 64) * 2, stream>>>(
      xcat, 672, 32, idx2, c2w1p, c2b1, 64, 64, c2w2p, c2b2, 128, xcat + 32, 672, N);

  // --- EdgeConv 3: x2(C=128) -> xcat[:, 160:672]
  knn_topk<<<dim3(N, B), 32, knn_smem(128), stream>>>(xcat + 32, 672, 128, N, idx3);
  edgeconv_wmma<<<dim3(N / 2, B), 32, (size_t)(16 * 256 + 16 * 256) * 2, stream>>>(
      xcat + 32, 672, 128, idx3, c3w1p, c3b1, 256, 256, c3w2p, c3b2, 512,
      xcat + 160, 672, N);

  // --- sf = MLP(672->256->128)(xcat)
  mlp2_wmma<<<(B * N) / 16, 32, (size_t)(16 * 672 + 16 * 256) * 2, stream>>>(
      xcat, 672, 672, smw1p, smb1, 256, 256, smw2p, smb2, 128, sf, 128);

  // --- global max pool, hi/hj/hg
  maxpool_kernel<<<B, 128, 0, stream>>>(sf, g, N);
  gemm_wmma<<<dim3((B * N) / 16, 8), 32, (size_t)16 * 128 * 2, stream>>>(
      sf, 128, 128, ehiw, 128, hi, 128);
  gemm_wmma<<<dim3((B * N) / 16, 8), 32, (size_t)16 * 128 * 2, stream>>>(
      sf, 128, 128, ehjw, 128, hj, 128);
  hg_kernel<<<1, 256, 0, stream>>>(g, ecw1, ecb1, hg);

  // --- fused pair classifier
  pair_kernel<<<N, 256, 0, stream>>>(hi, hj, hg, ecw2, ecb2, out, N);
}
